// CausalSelfAttention_56925496541402
// MI455X (gfx1250) — matrix operand order repaired and checked by reading, not compiled, not run on hardware
//
#include <hip/hip_runtime.h>
#include <hip/hip_bf16.h>

typedef __bf16 bf16_t;
typedef __attribute__((ext_vector_type(16))) __bf16 v16bf;
typedef __attribute__((ext_vector_type(8)))  float  v8f;

#define N_HEAD 8
#define HEAD_DIM 128
#define N_EMBD 1024

// ---------------------------------------------------------------------------
// elementwise f32 -> bf16
__global__ void cvt_bf16_kernel(const float* __restrict__ in, bf16_t* __restrict__ out, int n) {
    int i = blockIdx.x * 256 + threadIdx.x;
    if (i < n) out[i] = (bf16_t)in[i];
}

// transpose-convert weight W[K][N] f32 -> Wt[N][K] bf16, LDS-tiled (coalesced both ways)
__global__ __launch_bounds__(256) void transpose_cvt_kernel(const float* __restrict__ W,
                                                            bf16_t* __restrict__ Wt,
                                                            int K, int N) {
    __shared__ float tile[16][17];
    const int tx = threadIdx.x, ty = threadIdx.y;
    const int nb = blockIdx.x * 16, kb = blockIdx.y * 16;
    tile[ty][tx] = W[(size_t)(kb + ty) * N + nb + tx];
    __syncthreads();
    Wt[(size_t)(nb + ty) * K + kb + tx] = (bf16_t)tile[tx][ty];
}

// ---------------------------------------------------------------------------
// GEMM: C[M][N] = A[M][K] * Bt[N][K]^T, bf16 in, f32 WMMA accum.
// Register tiling: each wave computes 32 rows x 64 cols (2 M-tiles x 4 N-tiles,
// 8 accumulators); A frags reused 4x, B frags reused 2x (~21 flops/byte).
// 8 waves/block -> block covers 256 rows x 64 cols.
template <int STORE_BF16>
__global__ __launch_bounds__(256) void gemm_wmma_bf16(const bf16_t* __restrict__ A,
                                                      const bf16_t* __restrict__ Bt,
                                                      void* __restrict__ C,
                                                      int M, int N, int K) {
    const int wave = threadIdx.x >> 5;
    const int lane = threadIdx.x & 31;
    const int l15  = lane & 15;
    const int hi   = lane >> 4;
    const int m0   = blockIdx.x * 256 + wave * 32;
    const int n0   = blockIdx.y * 64;
    if (m0 >= M) return;

    v8f acc[2][4];
    const v8f vz = {};
#pragma unroll
    for (int mi = 0; mi < 2; ++mi)
#pragma unroll
        for (int j = 0; j < 4; ++j) acc[mi][j] = vz;

    const bf16_t* arow0 = A + (size_t)(m0 + l15) * K + hi * 16;      // A frag rows m0..m0+15
    const bf16_t* arow1 = arow0 + (size_t)16 * K;                    // rows m0+16..m0+31
    const bf16_t* brow  = Bt + (size_t)(n0 + l15) * K + hi * 16;     // B frag cols n0..n0+15 (+16j)

    for (int k0 = 0; k0 < K; k0 += 32) {
        // prefetch next chunk (global_prefetch_b8)
        if (k0 + 64 < K) {
            __builtin_prefetch(arow0 + k0 + 64, 0, 0);
            __builtin_prefetch(arow1 + k0 + 64, 0, 0);
            __builtin_prefetch(brow + k0 + 64, 0, 0);
        }
        v16bf a0, a1, bfv[4];
#pragma unroll
        for (int i = 0; i < 8; ++i) {
            a0[i] = arow0[k0 + i];     a0[8 + i] = arow0[k0 + 8 + i];
            a1[i] = arow1[k0 + i];     a1[8 + i] = arow1[k0 + 8 + i];
        }
#pragma unroll
        for (int j = 0; j < 4; ++j) {
            const bf16_t* bp = brow + (size_t)(j * 16) * K + k0;
#pragma unroll
            for (int i = 0; i < 16; ++i) bfv[j][i] = bp[i];
        }
#pragma unroll
        for (int j = 0; j < 4; ++j) {
            acc[0][j] = __builtin_amdgcn_wmma_f32_16x16x32_bf16(false, a0, false, bfv[j], (short)0,
                                                                acc[0][j], false, false);
            acc[1][j] = __builtin_amdgcn_wmma_f32_16x16x32_bf16(false, a1, false, bfv[j], (short)0,
                                                                acc[1][j], false, false);
        }
    }

#pragma unroll
    for (int mi = 0; mi < 2; ++mi) {
#pragma unroll
        for (int j = 0; j < 4; ++j) {
            const size_t colbase = (size_t)(n0 + j * 16 + l15);
            if (STORE_BF16) {
                bf16_t* Cb = (bf16_t*)C;
#pragma unroll
                for (int r = 0; r < 8; ++r)
                    Cb[(size_t)(m0 + mi * 16 + r + 8 * hi) * N + colbase] = (bf16_t)acc[mi][j][r];
            } else {
                float* Cf = (float*)C;
#pragma unroll
                for (int r = 0; r < 8; ++r)
                    Cf[(size_t)(m0 + mi * 16 + r + 8 * hi) * N + colbase] = acc[mi][j][r];
            }
        }
    }
}

// ---------------------------------------------------------------------------
// Fused: gate = 2*sigmoid(x[:, :32] @ w_gate); v += gate * ve (-> vt [h][128][T+32]);
// q,k: RoPE + RMS-norm -> qn/kn [h][T+32][128] bf16. One wave per (t, h).
__global__ __launch_bounds__(256) void fuse_gate_rope_norm(
    const bf16_t* __restrict__ qb, const bf16_t* __restrict__ kb, const bf16_t* __restrict__ vb,
    const float* __restrict__ x, const float* __restrict__ ve,
    const float* __restrict__ cosp, const float* __restrict__ sinp,
    const float* __restrict__ wgate,
    bf16_t* __restrict__ qn, bf16_t* __restrict__ kn, bf16_t* __restrict__ vt, int T) {
    const int wave = threadIdx.x >> 5;
    const int lane = threadIdx.x & 31;
    const int gw = blockIdx.x * 8 + wave;
    const int t = gw >> 3;
    const int h = gw & 7;
    if (t >= T) return;
    const int TP = T + 32;
    const size_t base = (size_t)t * N_EMBD + h * HEAD_DIM;

    // gate: dot over 32 channels, wave-wide reduce
    float gp = x[(size_t)t * N_EMBD + lane] * wgate[lane * N_HEAD + h];
#pragma unroll
    for (int mk = 16; mk > 0; mk >>= 1) gp += __shfl_xor(gp, mk, 32);
    const float gate = 2.0f / (1.0f + __expf(-gp));

    // v + gate*ve, store transposed [h][d][t]
#pragma unroll
    for (int i = 0; i < 4; ++i) {
        const int d = lane + 32 * i;
        float v = (float)vb[base + d] + gate * ve[base + d];
        vt[((size_t)h * HEAD_DIM + d) * TP + t] = (bf16_t)v;
    }

    const float c0 = cosp[t * 64 + lane],      s0 = sinp[t * 64 + lane];
    const float c1 = cosp[t * 64 + lane + 32], s1 = sinp[t * 64 + lane + 32];

    // ---- q: rope + rmsnorm ----
    {
        float x0 = (float)qb[base + lane],      x1 = (float)qb[base + lane + 32];
        float x2 = (float)qb[base + lane + 64], x3 = (float)qb[base + lane + 96];
        float r0 = x0 * c0 + x2 * s0, r1 = x1 * c1 + x3 * s1;
        float r2 = -x0 * s0 + x2 * c0, r3 = -x1 * s1 + x3 * c1;
        float ss = r0 * r0 + r1 * r1 + r2 * r2 + r3 * r3;
#pragma unroll
        for (int mk = 16; mk > 0; mk >>= 1) ss += __shfl_xor(ss, mk, 32);
        float sc = rsqrtf(ss * (1.0f / 128.0f) + 1e-6f);
        size_t ob = ((size_t)h * TP + t) * HEAD_DIM;
        qn[ob + lane]      = (bf16_t)(r0 * sc);
        qn[ob + lane + 32] = (bf16_t)(r1 * sc);
        qn[ob + lane + 64] = (bf16_t)(r2 * sc);
        qn[ob + lane + 96] = (bf16_t)(r3 * sc);
    }
    // ---- k: rope + rmsnorm ----
    {
        float x0 = (float)kb[base + lane],      x1 = (float)kb[base + lane + 32];
        float x2 = (float)kb[base + lane + 64], x3 = (float)kb[base + lane + 96];
        float r0 = x0 * c0 + x2 * s0, r1 = x1 * c1 + x3 * s1;
        float r2 = -x0 * s0 + x2 * c0, r3 = -x1 * s1 + x3 * c1;
        float ss = r0 * r0 + r1 * r1 + r2 * r2 + r3 * r3;
#pragma unroll
        for (int mk = 16; mk > 0; mk >>= 1) ss += __shfl_xor(ss, mk, 32);
        float sc = rsqrtf(ss * (1.0f / 128.0f) + 1e-6f);
        size_t ob = ((size_t)h * TP + t) * HEAD_DIM;
        kn[ob + lane]      = (bf16_t)(r0 * sc);
        kn[ob + lane + 32] = (bf16_t)(r1 * sc);
        kn[ob + lane + 64] = (bf16_t)(r2 * sc);
        kn[ob + lane + 96] = (bf16_t)(r3 * sc);
    }
}

// ---------------------------------------------------------------------------
// Flash attention, sliding window. 1 wave per (16-query tile, head).
__global__ __launch_bounds__(32) void attn_wmma_kernel(
    const bf16_t* __restrict__ qn, const bf16_t* __restrict__ kn, const bf16_t* __restrict__ vt,
    const int* __restrict__ winp, bf16_t* __restrict__ ybf, int T) {
    __shared__ bf16_t sP[16 * 32];
    const int TP = T + 32;
    const int lane = threadIdx.x & 31;
    const int l15 = lane & 15;
    const int hi = lane >> 4;
    const int q0 = blockIdx.x * 16;
    const int h = blockIdx.y;
    const int window = winp[0];
    const float scale = 0.08838834764831845f;  // 1/sqrt(128)
    const float NEGBIG = -1e30f;

    // Q fragments (16x128 = 4 x (16x32)) kept in registers
    const bf16_t* qrow = qn + ((size_t)h * TP + q0 + l15) * HEAD_DIM + hi * 16;
    v16bf qf[4];
#pragma unroll
    for (int kbk = 0; kbk < 4; ++kbk) {
#pragma unroll
        for (int i = 0; i < 8; ++i) {
            qf[kbk][i] = qrow[kbk * 32 + i];
            qf[kbk][8 + i] = qrow[kbk * 32 + 8 + i];
        }
    }

    float m[8], lsum[8];
    v8f o[8];
    const v8f vzero = {};
#pragma unroll
    for (int r = 0; r < 8; ++r) { m[r] = NEGBIG; lsum[r] = 0.f; }
#pragma unroll
    for (int t = 0; t < 8; ++t) o[t] = vzero;

    int jlo = q0 - window + 1;
    if (jlo < 0) jlo = 0;
    jlo &= ~31;

    for (int j0 = jlo; j0 <= q0 + 15; j0 += 32) {
        const bf16_t* krow = kn + ((size_t)h * TP + j0 + l15) * HEAD_DIM + hi * 16;
        // prefetch next key block (K rows and V columns)
        __builtin_prefetch(krow + 32 * HEAD_DIM, 0, 0);

        // S = Q K^T for 32 keys (two 16x16 N-tiles)
        v8f s0 = {}, s1 = {};
#pragma unroll
        for (int kbk = 0; kbk < 4; ++kbk) {
            v16bf b0, b1;
            const bf16_t* kr0 = krow + kbk * 32;
            const bf16_t* kr1 = kr0 + 16 * HEAD_DIM;
#pragma unroll
            for (int i = 0; i < 16; ++i) { b0[i] = kr0[i]; b1[i] = kr1[i]; }
            s0 = __builtin_amdgcn_wmma_f32_16x16x32_bf16(false, qf[kbk], false, b0, (short)0, s0,
                                                         false, false);
            s1 = __builtin_amdgcn_wmma_f32_16x16x32_bf16(false, qf[kbk], false, b1, (short)0, s1,
                                                         false, false);
        }
        // online softmax (per row: 16 lanes of the half-wave hold the 16+16 cols)
#pragma unroll
        for (int r = 0; r < 8; ++r) {
            const int row = q0 + r + 8 * hi;
            const int c0i = j0 + l15, c1i = c0i + 16;
            float e0 = s0[r] * scale, e1 = s1[r] * scale;
            e0 = ((row >= c0i) && (row - c0i < window)) ? e0 : NEGBIG;
            e1 = ((row >= c1i) && (row - c1i < window)) ? e1 : NEGBIG;
            float bm = fmaxf(e0, e1);
            bm = fmaxf(bm, __shfl_xor(bm, 1, 32));
            bm = fmaxf(bm, __shfl_xor(bm, 2, 32));
            bm = fmaxf(bm, __shfl_xor(bm, 4, 32));
            bm = fmaxf(bm, __shfl_xor(bm, 8, 32));
            float mnew = fmaxf(m[r], bm);
            float corr = __expf(m[r] - mnew);
            m[r] = mnew;
            float p0 = __expf(e0 - mnew), p1 = __expf(e1 - mnew);
            float rs = p0 + p1;
            rs += __shfl_xor(rs, 1, 32);
            rs += __shfl_xor(rs, 2, 32);
            rs += __shfl_xor(rs, 4, 32);
            rs += __shfl_xor(rs, 8, 32);
            lsum[r] = lsum[r] * corr + rs;
#pragma unroll
            for (int t = 0; t < 8; ++t) o[t][r] *= corr;
            s0[r] = p0;
            s1[r] = p1;
        }
        // transpose P (C-layout) -> A-fragment via LDS (single-wave workgroup)
        __syncthreads();
#pragma unroll
        for (int r = 0; r < 8; ++r) {
            sP[(r + 8 * hi) * 32 + l15] = (bf16_t)s0[r];
            sP[(r + 8 * hi) * 32 + 16 + l15] = (bf16_t)s1[r];
        }
        __syncthreads();
        v16bf pf;
#pragma unroll
        for (int i = 0; i < 8; ++i) {
            pf[i] = sP[l15 * 32 + hi * 16 + i];
            pf[8 + i] = sP[l15 * 32 + hi * 16 + 8 + i];
        }
        // O += P @ V  (V pre-transposed [h][d][T])
#pragma unroll
        for (int t = 0; t < 8; ++t) {
            v16bf vf;
            const bf16_t* vr = vt + ((size_t)h * HEAD_DIM + t * 16 + l15) * TP + j0 + hi * 16;
            __builtin_prefetch(vr + 32, 0, 0);
#pragma unroll
            for (int i = 0; i < 16; ++i) vf[i] = vr[i];
            o[t] = __builtin_amdgcn_wmma_f32_16x16x32_bf16(false, pf, false, vf, (short)0, o[t],
                                                           false, false);
        }
    }

    // finalize: divide by row sums, write y[t][h*128+d] bf16
#pragma unroll
    for (int r = 0; r < 8; ++r) {
        const float inv = 1.0f / lsum[r];
        const size_t row = q0 + r + 8 * hi;
#pragma unroll
        for (int t = 0; t < 8; ++t)
            ybf[row * N_EMBD + (size_t)h * HEAD_DIM + t * 16 + l15] = (bf16_t)(o[t][r] * inv);
    }
}

// ---------------------------------------------------------------------------
extern "C" void kernel_launch(void* const* d_in, const int* in_sizes, int n_in,
                              void* d_out, int out_size, void* d_ws, size_t ws_size,
                              hipStream_t stream) {
    const float* x     = (const float*)d_in[0];
    const float* ve    = (const float*)d_in[1];
    const float* cosp  = (const float*)d_in[2];
    const float* sinp  = (const float*)d_in[3];
    const float* wq    = (const float*)d_in[4];
    const float* wk    = (const float*)d_in[5];
    const float* wv    = (const float*)d_in[6];
    const float* wgate = (const float*)d_in[7];
    const float* wproj = (const float*)d_in[8];
    const int*   winp  = (const int*)d_in[9];
    float* out = (float*)d_out;

    const int T = 4096, C = N_EMBD;
    const int TP = T + 32;

    char* ws = (char*)d_ws;
    size_t off = 0;
    auto alloc = [&](size_t b) -> void* {
        void* p = ws + off;
        off += (b + 255) & ~(size_t)255;
        return p;
    };
    bf16_t* xb  = (bf16_t*)alloc((size_t)T * C * 2);
    bf16_t* wqT = (bf16_t*)alloc((size_t)C * C * 2);
    bf16_t* wkT = (bf16_t*)alloc((size_t)C * C * 2);
    bf16_t* wvT = (bf16_t*)alloc((size_t)C * C * 2);
    bf16_t* wpT = (bf16_t*)alloc((size_t)C * C * 2);
    bf16_t* qb  = (bf16_t*)alloc((size_t)T * C * 2);
    bf16_t* kb  = (bf16_t*)alloc((size_t)T * C * 2);
    bf16_t* vb  = (bf16_t*)alloc((size_t)T * C * 2);
    bf16_t* qn  = (bf16_t*)alloc((size_t)N_HEAD * TP * HEAD_DIM * 2);
    bf16_t* knb = (bf16_t*)alloc((size_t)N_HEAD * TP * HEAD_DIM * 2);
    bf16_t* vtb = (bf16_t*)alloc((size_t)N_HEAD * HEAD_DIM * TP * 2);
    bf16_t* yb  = (bf16_t*)alloc((size_t)T * C * 2);

    // 1) conversions
    cvt_bf16_kernel<<<(T * C + 255) / 256, 256, 0, stream>>>(x, xb, T * C);
    dim3 tb(16, 16), tg(C / 16, C / 16);
    transpose_cvt_kernel<<<tg, tb, 0, stream>>>(wq, wqT, C, C);
    transpose_cvt_kernel<<<tg, tb, 0, stream>>>(wk, wkT, C, C);
    transpose_cvt_kernel<<<tg, tb, 0, stream>>>(wv, wvT, C, C);
    transpose_cvt_kernel<<<tg, tb, 0, stream>>>(wproj, wpT, C, C);

    // 2) QKV projections (WMMA), 256x64 per block
    dim3 gg(T / 256, C / 64);
    gemm_wmma_bf16<1><<<gg, 256, 0, stream>>>(xb, wqT, qb, T, C, C);
    gemm_wmma_bf16<1><<<gg, 256, 0, stream>>>(xb, wkT, kb, T, C, C);
    gemm_wmma_bf16<1><<<gg, 256, 0, stream>>>(xb, wvT, vb, T, C, C);

    // 3) gate + ve add + RoPE + RMS-norm (+ V transpose)
    fuse_gate_rope_norm<<<(T * N_HEAD) / 8, 256, 0, stream>>>(qb, kb, vb, x, ve, cosp, sinp, wgate,
                                                              qn, knb, vtb, T);

    // 4) sliding-window flash attention (WMMA)
    attn_wmma_kernel<<<dim3(T / 16, N_HEAD), 32, 0, stream>>>(qn, knb, vtb, winp, yb, T);

    // 5) output projection (WMMA) -> fp32 out
    gemm_wmma_bf16<0><<<gg, 256, 0, stream>>>(yb, wpT, out, T, C, C);
}
